// EdgeConvGRUCell_40346922778939
// MI455X (gfx1250) — matrix operand order repaired and checked by reading, not compiled, not run on hardware
//
#include <hip/hip_runtime.h>
#include <hip/hip_bf16.h>

typedef __attribute__((ext_vector_type(16))) _Float16 v16h;
typedef __attribute__((ext_vector_type(8)))  _Float16 v8h;
typedef __attribute__((ext_vector_type(4)))  _Float16 v4h;
typedef __attribute__((ext_vector_type(8)))  float    v8f;

// ---------------------------------------------------------------------------
// Kernel 1: build comb1 = concat(x, h) and initialize segment-max buffers.
// ---------------------------------------------------------------------------
__global__ void init_kernel(const float* __restrict__ x, const float* __restrict__ h,
                            float* __restrict__ comb1, float* __restrict__ gmax_g,
                            float* __restrict__ gmax_c, int N) {
    int i = blockIdx.x * blockDim.x + threadIdx.x;
    if (i >= N * 64) return;
    int node = i >> 6;
    int c    = i & 63;
    comb1[i]  = (c < 32) ? x[node * 32 + c] : h[node * 32 + (c - 32)];
    gmax_g[i] = -__builtin_inff();
    if (c < 32) gmax_c[node * 32 + c] = -__builtin_inff();
}

// ---------------------------------------------------------------------------
// Kernel 2/4: EdgeConv message GEMM + segment-max, one wave per block.
// Tile: 16 edges x 128 features (f16 A) times W (128 x NCOLS, f16 B frags in
// registers), f32 accumulate, then global_atomic_max_num_f32 scatter by dst.
// ---------------------------------------------------------------------------
template <int NCOLS>
__global__ __launch_bounds__(32)
void edgeconv_wmma_kernel(const float* __restrict__ feat,   // [N,64] fp32
                          const int*   __restrict__ eidx,   // [2,E]  int32
                          const float* __restrict__ W,      // [128, NCOLS]
                          const float* __restrict__ bias,   // [NCOLS]
                          float*       __restrict__ gmax,   // [N, NCOLS]
                          int nEdges) {
    constexpr int NT = NCOLS / 16;   // N tiles (4 for gate, 2 for candidate)
    constexpr int KT = 4;            // K chunks of 32 (K = 128)

    __shared__ _Float16 mtile[16][128];  // 16 edges x 128 f16 message features
    __shared__ int      sdst[16];
    __shared__ int      svalid[16];

    const int ln     = threadIdx.x;      // 0..31
    const int lane15 = ln & 15;
    const int hi     = (ln >> 4) & 1;    // half-wave id

    // ---- Load B fragments (weights) into registers once per block ----
    // B frags use contiguous k here: lanes 0-15 hold K=0..15 (lane = column N),
    //                     lanes 16-31 hold K=16..31. Element j -> K = base+j.
    v16h bfrag[KT][NT];
#pragma unroll
    for (int kk = 0; kk < KT; ++kk) {
#pragma unroll
        for (int nn = 0; nn < NT; ++nn) {
            const int kbase = kk * 32 + hi * 16;
            const int col   = nn * 16 + lane15;
            v16h f;
#pragma unroll
            for (int j = 0; j < 16; ++j)
                f[j] = (_Float16)W[(kbase + j) * NCOLS + col];
            bfrag[kk][nn] = f;
        }
    }
    float biasv[NT];
#pragma unroll
    for (int nn = 0; nn < NT; ++nn) biasv[nn] = bias[nn * 16 + lane15];

    const int ntiles = (nEdges + 15) >> 4;
    for (int tile = blockIdx.x; tile < ntiles; tile += gridDim.x) {
        const int  e   = tile * 16 + lane15;
        const bool val = (e < nEdges);
        const int  si  = val ? eidx[e] : 0;           // source j
        const int  di  = val ? eidx[nEdges + e] : 0;  // dest   i

        if (hi == 0) { sdst[lane15] = di; svalid[lane15] = val ? 1 : 0; }

        // ---- Build the 16x128 f16 message tile in LDS ----
        const float* rowd = feat + (size_t)di * 64;
        if (hi == 0) {
            // k = 0..63 : x_i
#pragma unroll
            for (int k = 0; k < 64; k += 4) {
                float4 v = *(const float4*)(rowd + k);
                v4h pk = {(_Float16)v.x, (_Float16)v.y, (_Float16)v.z, (_Float16)v.w};
                *(v4h*)&mtile[lane15][k] = pk;
            }
        } else {
            // k = 64..127 : x_j - x_i
            const float* rows = feat + (size_t)si * 64;
#pragma unroll
            for (int k = 0; k < 64; k += 4) {
                float4 a = *(const float4*)(rows + k);
                float4 b = *(const float4*)(rowd + k);
                v4h pk = {(_Float16)(a.x - b.x), (_Float16)(a.y - b.y),
                          (_Float16)(a.z - b.z), (_Float16)(a.w - b.w)};
                *(v4h*)&mtile[lane15][64 + k] = pk;
            }
        }
        __syncthreads();

        // ---- WMMA: acc[nn] += A(kk) x B(kk,nn) ----
        v8f acc[NT] = {};
#pragma unroll
        for (int kk = 0; kk < KT; ++kk) {
            // A frag, same contiguous k order as B: lane<16 -> K 0..15; lane>=16 -> K 16..31
            const _Float16* p = &mtile[lane15][kk * 32 + hi * 16];
            v8h lo = *(const v8h*)p;
            v8h hv = *(const v8h*)(p + 8);
            v16h a = __builtin_shufflevector(lo, hv, 0, 1, 2, 3, 4, 5, 6, 7,
                                                     8, 9, 10, 11, 12, 13, 14, 15);
#pragma unroll
            for (int nn = 0; nn < NT; ++nn)
                acc[nn] = __builtin_amdgcn_wmma_f32_16x16x32_f16(
                    false, a, false, bfrag[kk][nn], (short)0, acc[nn], false, false);
        }
        __syncthreads();

        // ---- Segment-max scatter: D layout lane<16 -> M=r, lane>=16 -> M=8+r ----
#pragma unroll
        for (int nn = 0; nn < NT; ++nn) {
            const int col = nn * 16 + lane15;
#pragma unroll
            for (int r = 0; r < 8; ++r) {
                const int mrow = r + hi * 8;
                if (svalid[mrow]) {
                    float vv = acc[nn][r] + biasv[nn];
                    float* p = &gmax[(size_t)sdst[mrow] * NCOLS + col];
                    asm volatile("global_atomic_max_num_f32 %0, %1, off"
                                 :
                                 : "v"(p), "v"(vv)
                                 : "memory");
                }
            }
        }
    }
}

// ---------------------------------------------------------------------------
// Kernel 3: sigmoid gates, build comb2 = concat(x, h*reset), stash update.
// ---------------------------------------------------------------------------
__global__ void gate_finalize_kernel(const float* __restrict__ x, const float* __restrict__ h,
                                     const float* __restrict__ gmax_g,
                                     float* __restrict__ comb2, float* __restrict__ upd, int N) {
    int i = blockIdx.x * blockDim.x + threadIdx.x;
    if (i >= N * 32) return;
    int node = i >> 5;
    int c    = i & 31;
    float r = gmax_g[node * 64 + c];
    float u = gmax_g[node * 64 + 32 + c];
    if (!__builtin_isfinite(r)) r = 0.0f;   // nodes with no incoming edges -> 0
    if (!__builtin_isfinite(u)) u = 0.0f;
    float rg = 1.0f / (1.0f + __expf(-r));
    float ug = 1.0f / (1.0f + __expf(-u));
    comb2[node * 64 + c]      = x[i];
    comb2[node * 64 + 32 + c] = h[i] * rg;
    upd[i] = ug;
}

// ---------------------------------------------------------------------------
// Kernel 5: h_next = (1-u)*h + u*tanh(candidate_max)
// ---------------------------------------------------------------------------
__global__ void final_kernel(const float* __restrict__ h, const float* __restrict__ gmax_c,
                             const float* __restrict__ upd, float* __restrict__ out, int N) {
    int i = blockIdx.x * blockDim.x + threadIdx.x;
    if (i >= N * 32) return;
    float v = gmax_c[i];
    if (!__builtin_isfinite(v)) v = 0.0f;
    float ht = tanhf(v);
    float u  = upd[i];
    out[i] = (1.0f - u) * h[i] + u * ht;
}

// ---------------------------------------------------------------------------
extern "C" void kernel_launch(void* const* d_in, const int* in_sizes, int n_in,
                              void* d_out, int out_size, void* d_ws, size_t ws_size,
                              hipStream_t stream) {
    const float* h  = (const float*)d_in[0];
    const float* x  = (const float*)d_in[1];
    const int*   eg = (const int*)d_in[2];
    const int*   ec = (const int*)d_in[3];
    const float* Wg = (const float*)d_in[4];
    const float* bg = (const float*)d_in[5];
    const float* Wc = (const float*)d_in[6];
    const float* bc = (const float*)d_in[7];
    float* out = (float*)d_out;

    const int N = in_sizes[0] / 32;   // h is [N,32]
    const int E = in_sizes[2] / 2;    // edge_index is [2,E]

    float* ws = (float*)d_ws;
    size_t o = 0;
    float* comb1 = ws + o; o += (size_t)N * 64;
    float* comb2 = ws + o; o += (size_t)N * 64;
    float* gmaxg = ws + o; o += (size_t)N * 64;
    float* gmaxc = ws + o; o += (size_t)N * 32;
    float* upd   = ws + o; o += (size_t)N * 32;
    (void)ws_size; (void)n_in; (void)out_size;

    const int ntiles  = (E + 15) / 16;
    const int eblocks = ntiles < 4096 ? ntiles : 4096;

    init_kernel<<<(N * 64 + 255) / 256, 256, 0, stream>>>(x, h, comb1, gmaxg, gmaxc, N);
    edgeconv_wmma_kernel<64><<<eblocks, 32, 0, stream>>>(comb1, eg, Wg, bg, gmaxg, E);
    gate_finalize_kernel<<<(N * 32 + 255) / 256, 256, 0, stream>>>(x, h, gmaxg, comb2, upd, N);
    edgeconv_wmma_kernel<32><<<eblocks, 32, 0, stream>>>(comb2, ec, Wc, bc, gmaxc, E);
    final_kernel<<<(N * 32 + 255) / 256, 256, 0, stream>>>(h, gmaxc, upd, out, N);
}
